// GCNEncoder_32315333935770
// MI455X (gfx1250) — matrix operand order repaired and checked by reading, not compiled, not run on hardware
//
#include <hip/hip_runtime.h>

typedef unsigned int u32;
typedef __attribute__((ext_vector_type(16))) __bf16 v16bf;
typedef __attribute__((ext_vector_type(8)))  float  v8f;

#define N_NODES 50000
#define N_EDGES 320000
#define IN_F    256
#define HID     256
#define OUT_F   128
#define MT      32      // rows per block (2x 16-row tiles per wave)

// ---------- helpers ----------
static __device__ __forceinline__ u32 pack2_bf16(float a, float b) {
    // round-to-nearest-even fp32 -> bf16, pack low|high
    u32 ua = __float_as_uint(a);
    u32 ub = __float_as_uint(b);
    ua += 0x7FFFu + ((ua >> 16) & 1u);
    ub += 0x7FFFu + ((ub >> 16) & 1u);
    return (ua >> 16) | (ub & 0xFFFF0000u);
}

// ---------- degree / norm ----------
__global__ void k_deg_init(float* __restrict__ deg, int n) {
    int i = blockIdx.x * blockDim.x + threadIdx.x;
    if (i < n) deg[i] = 1.0f;  // self loop
}

__global__ void k_deg_count(const int* __restrict__ ei, float* __restrict__ deg, int e) {
    int i = blockIdx.x * blockDim.x + threadIdx.x;
    if (i < e) atomicAdd(&deg[ei[i]], 1.0f);   // ei[0..E) = destinations (row)
}

__global__ void k_dinv(float* __restrict__ deg, int n) {
    int i = blockIdx.x * blockDim.x + threadIdx.x;
    if (i < n) deg[i] = rsqrtf(deg[i]);        // deg >= 1 always
}

// ---------- fp32 -> packed bf16 (pairs along fastest dim) ----------
__global__ void k_cvt_pack(const float* __restrict__ src, u32* __restrict__ dst, long npair) {
    long i = (long)blockIdx.x * blockDim.x + threadIdx.x;
    if (i < npair) dst[i] = pack2_bf16(src[2 * i], src[2 * i + 1]);
}

// W [K][Nc] fp32 -> fragment-major bf16: Bf[ntile][kstep][lane][v], v=0..7 dwords.
// lane: hi=lane>>4 selects K offset (+8), lo=lane&15 is the output column n.
// dword v holds K = kstep*32 + (v>>2)*16 + hi*8 + 2*(v&3) (+1).
__global__ void k_pack_w_frag(const float* __restrict__ W, u32* __restrict__ Bf,
                              int K, int Nc) {
    int i = blockIdx.x * blockDim.x + threadIdx.x;
    int ksteps = K >> 5;
    int tot = (Nc / 16) * ksteps * 256;          // 32 lanes * 8 dwords per (ntile,kstep)
    if (i >= tot) return;
    int v     = i & 7;
    int l     = (i >> 3) & 31;
    int kt    = (i >> 8) % ksteps;
    int ntile = (i >> 8) / ksteps;
    int hi = l >> 4, lo = l & 15;
    int k = kt * 32 + (v >> 2) * 16 + hi * 8 + 2 * (v & 3);
    int n = ntile * 16 + lo;
    Bf[i] = pack2_bf16(W[k * Nc + n], W[(k + 1) * Nc + n]);
}

// ---------- bf16 WMMA GEMM with fused self-loop epilogue ----------
//   H[m][n] = sum_k A[m][k] * W[k][n]                  (raw, for edge gathers)
//   O[m][n] = H[m][n] * dinv[m]^2 (+ bias[n])          (scatter accumulator init)
// A : [M][K/2] dwords (row-major bf16 pairs)
// Bf: fragment-major (see k_pack_w_frag)
__global__ __launch_bounds__(256) void k_gemm_bf16(const u32* __restrict__ A,
                                                   const u32* __restrict__ Bf,
                                                   const float* __restrict__ dinv,
                                                   const float* __restrict__ bias,
                                                   float* __restrict__ H,
                                                   float* __restrict__ O,
                                                   int M, int K, int Nc) {
    const int lane  = threadIdx.x & 31;
    const int wave  = threadIdx.x >> 5;
    const int r0    = blockIdx.x * MT;
    const int ntile = blockIdx.y * 8 + wave;
    const int n0    = ntile * 16;
    const int Kp    = K >> 1;                 // dwords per A row
    const int hi    = lane >> 4;
    const int lo    = lane & 15;
    const int ksteps = K >> 5;

    int ra = r0 + lo;        if (ra > M - 1) ra = M - 1;   // clamp (stores are guarded)
    int rb = r0 + 16 + lo;   if (rb > M - 1) rb = M - 1;

    const u32* aRow0 = A + (size_t)ra * Kp + hi * 4;
    const u32* aRow1 = A + (size_t)rb * Kp + hi * 4;
    const u32* bp    = Bf + ((size_t)ntile * ksteps * 32 + lane) * 8;

    v8f acc0 = {}, acc1 = {};
#pragma unroll
    for (int kt = 0; kt < 8; ++kt) {          // K/32 == 8 for both layers
        union { uint4 q[2]; v16bf v; } a0, a1, b;
        b.q[0]  = *(const uint4*)(bp + (size_t)kt * 256);
        b.q[1]  = *(const uint4*)(bp + (size_t)kt * 256 + 4);
        a0.q[0] = *(const uint4*)(aRow0 + kt * 16);
        a0.q[1] = *(const uint4*)(aRow0 + kt * 16 + 8);
        a1.q[0] = *(const uint4*)(aRow1 + kt * 16);
        a1.q[1] = *(const uint4*)(aRow1 + kt * 16 + 8);
        acc0 = __builtin_amdgcn_wmma_f32_16x16x32_bf16(
            false, a0.v, false, b.v, (short)0, acc0, false, false);
        acc1 = __builtin_amdgcn_wmma_f32_16x16x32_bf16(
            false, a1.v, false, b.v, (short)0, acc1, false, false);
    }

    // epilogue: lane lo = column, lane group hi -> +8 rows, VGPR j -> row
    float bv = bias ? bias[n0 + lo] : 0.0f;
#pragma unroll
    for (int t = 0; t < 2; ++t) {
        v8f acc = t ? acc1 : acc0;
        int mbase = r0 + t * 16 + hi * 8;
#pragma unroll
        for (int j = 0; j < 8; ++j) {
            int m = mbase + j;
            if (m < M) {
                float c = acc[j];
                size_t idx = (size_t)m * Nc + n0 + lo;
                float d = dinv[m];
                H[idx] = c;
                O[idx] = c * d * d + bv;
            }
        }
    }
}

// ---------- propagation: one wave per edge, O[row] += H[col]*dinv[row]*dinv[col] ----------
__global__ void k_edge_prop(const int* __restrict__ ei, int E,
                            const float* __restrict__ dinv,
                            const float* __restrict__ H,
                            float* __restrict__ O, int F) {
    int widx = (int)(((long)blockIdx.x * blockDim.x + threadIdx.x) >> 5);
    int lane = threadIdx.x & 31;
    if (widx >= E) return;
    int row = ei[widx];           // destination
    int col = ei[E + widx];       // source
    float w = dinv[row] * dinv[col];
    const float* hs = H + (size_t)col * F;
    float* od = O + (size_t)row * F;
#pragma unroll 4
    for (int f = lane; f < F; f += 32)
        atomicAdd(&od[f], hs[f] * w);
}

// a = relu(O1 + b1), repacked to bf16 pairs for layer-2 GEMM
__global__ void k_bias_relu_cvt(const float* __restrict__ O1, const float* __restrict__ b,
                                u32* __restrict__ Ab, int F, long npair) {
    long i = (long)blockIdx.x * blockDim.x + threadIdx.x;
    if (i >= npair) return;
    int f2 = (int)(i % (F / 2));
    float x0 = fmaxf(O1[2 * i]     + b[2 * f2],     0.0f);
    float x1 = fmaxf(O1[2 * i + 1] + b[2 * f2 + 1], 0.0f);
    Ab[i] = pack2_bf16(x0, x1);
}

// ---------- launcher ----------
extern "C" void kernel_launch(void* const* d_in, const int* in_sizes, int n_in,
                              void* d_out, int out_size, void* d_ws, size_t ws_size,
                              hipStream_t stream) {
    (void)in_sizes; (void)n_in; (void)out_size; (void)ws_size;
    const float* x  = (const float*)d_in[0];
    const int*   ei = (const int*)d_in[1];
    const float* W1 = (const float*)d_in[2];
    const float* b1 = (const float*)d_in[3];
    const float* W2 = (const float*)d_in[4];
    const float* b2 = (const float*)d_in[5];
    float* out = (float*)d_out;

    // workspace carve-up (256B aligned)
    char* ws = (char*)d_ws;
    size_t off = 0;
    auto carve = [&](size_t bytes) { void* p = ws + off; off += (bytes + 255) & ~(size_t)255; return p; };
    float* dinv = (float*)carve((size_t)N_NODES * 4);
    u32*   w1f  = (u32*)  carve((size_t)(IN_F / 2) * HID * 4);        // 128 KB fragment-major
    u32*   w2f  = (u32*)  carve((size_t)(HID / 2) * OUT_F * 4);       // 64 KB
    u32*   xb   = (u32*)  carve((size_t)N_NODES * (IN_F / 2) * 4);    // 25.6 MB (reused as a1b)
    float* h1   = (float*)carve((size_t)N_NODES * HID * 4);           // 51.2 MB (reused as h2)
    float* o1   = (float*)carve((size_t)N_NODES * HID * 4);           // 51.2 MB
    u32*   a1b  = xb;   // layer-2 bf16 activations reuse xb
    float* h2   = h1;   // layer-2 pre-prop reuses h1

    const int T = 256;
    // 1) degrees -> dinv
    k_deg_init <<<(N_NODES + T - 1) / T, T, 0, stream>>>(dinv, N_NODES);
    k_deg_count<<<(N_EDGES + T - 1) / T, T, 0, stream>>>(ei, dinv, N_EDGES);
    k_dinv     <<<(N_NODES + T - 1) / T, T, 0, stream>>>(dinv, N_NODES);
    // 2) pack inputs / weights to bf16
    {
        long np = (long)N_NODES * IN_F / 2;
        k_cvt_pack<<<(unsigned)((np + T - 1) / T), T, 0, stream>>>(x, xb, np);
        k_pack_w_frag<<<((IN_F / 2) * HID + T - 1) / T, T, 0, stream>>>(W1, w1f, IN_F, HID);
        k_pack_w_frag<<<((HID / 2) * OUT_F + T - 1) / T, T, 0, stream>>>(W2, w2f, HID, OUT_F);
    }
    // 3) GEMM1 (+ fused self-loop init of o1): h1 = xb@W1, o1 = h1*dinv^2
    k_gemm_bf16<<<dim3((N_NODES + MT - 1) / MT, HID / 128), T, 0, stream>>>(
        xb, w1f, dinv, nullptr, h1, o1, N_NODES, IN_F, HID);
    // 4) edge scatter layer 1, then bias+relu+bf16 repack
    k_edge_prop<<<(N_EDGES * 32 + T - 1) / T, T, 0, stream>>>(ei, N_EDGES, dinv, h1, o1, HID);
    {
        long np = (long)N_NODES * HID / 2;
        k_bias_relu_cvt<<<(unsigned)((np + T - 1) / T), T, 0, stream>>>(o1, b1, a1b, HID, np);
    }
    // 5) GEMM2 (+ fused self-loop + bias init of out): h2 = a1b@W2, out = h2*dinv^2 + b2
    k_gemm_bf16<<<dim3((N_NODES + MT - 1) / MT, OUT_F / 128), T, 0, stream>>>(
        a1b, w2f, dinv, b2, h2, out, N_NODES, HID, OUT_F);
    // 6) edge scatter layer 2 into d_out
    k_edge_prop<<<(N_EDGES * 32 + T - 1) / T, T, 0, stream>>>(ei, N_EDGES, dinv, h2, out, OUT_F);
}
